// SuperiorHybridGNN_89017492177272
// MI455X (gfx1250) — matrix operand order repaired and checked by reading, not compiled, not run on hardware
//
#include <hip/hip_runtime.h>
#include <math.h>

#define HDIM 128

typedef __attribute__((ext_vector_type(16))) _Float16 v16h;
typedef __attribute__((ext_vector_type(8)))  float    v8f;

__device__ __forceinline__ void atomAddF(float* p, float v) {
  __hip_atomic_fetch_add(p, v, __ATOMIC_RELAXED, __HIP_MEMORY_SCOPE_AGENT);
}
__device__ __forceinline__ void atomMaxU(unsigned* p, unsigned v) {
  __hip_atomic_fetch_max(p, v, __ATOMIC_RELAXED, __HIP_MEMORY_SCOPE_AGENT);
}
// order-preserving float<->uint encoding for atomic max
__device__ __forceinline__ unsigned encf(float f) {
  unsigned u = __float_as_uint(f);
  return (u & 0x80000000u) ? ~u : (u | 0x80000000u);
}
__device__ __forceinline__ float decf(unsigned u) {
  unsigned r = (u & 0x80000000u) ? (u & 0x7fffffffu) : ~u;
  return __uint_as_float(r);
}
// single v_exp_f32 + v_rcp_f32 instead of the IEEE div ladder
__device__ __forceinline__ float fast_sigmoid(float v) {
  return __builtin_amdgcn_rcpf(1.f + __expf(-v));
}

// ---------------------------------------------------------------------------
// WMMA GEMM: Y[r,m] = act( X[r,:K] . W[m,:K] + bias[m] (+ Y if ACC) )
// X row-major ldx, W row-major ldw ([out,in]), Y row-major ldy.
// K multiple of 32; nrows multiple of 16 (N=30000). One wave per 16x16 tile.
// Loads are unconditional (indices clamped, OOB columns masked by multiply) so
// the backend emits b128 clauses. ACT/ACC are template params -> straight-line
// epilogue, no per-element scalar branching.
// ACT: 0=none 1=relu 2=sigmoid
// ---------------------------------------------------------------------------
template <int ACT, int ACC>
__global__ void k_gemm_wmma(const float* __restrict__ X, int ldx,
                            const float* __restrict__ W, int ldw,
                            const float* __restrict__ bias,
                            float* __restrict__ Y, int ldy,
                            int nrows, int M, int K) {
  const int lane = threadIdx.x & 31;
  const int wave = threadIdx.x >> 5;
  const int nwaves = blockDim.x >> 5;
  const int row0 = blockIdx.x * 16;
  const int half = lane >> 4;         // 0: lanes 0-15, 1: lanes 16-31
  const int l16  = lane & 15;
  const int colTiles = (M + 15) >> 4;

  for (int ct = wave; ct < colTiles; ct += nwaves) {
    const int col0 = ct * 16;
    const int m    = col0 + l16;
    const int mcl  = (m < M) ? m : (M - 1);       // clamped, always valid
    const float fmask = (m < M) ? 1.f : 0.f;

    v8f c;
    {
      float base = (bias ? bias[mcl] : 0.f) * fmask;
#pragma unroll
      for (int i = 0; i < 8; ++i) {
        int r = row0 + half * 8 + i;
        float v0 = base;
        if (ACC) v0 += fmask * Y[(size_t)r * ldy + mcl];
        c[i] = v0;
      }
    }
    const int ar = row0 + l16;                     // always < nrows here
    for (int k0 = 0; k0 < K; k0 += 32) {
      // A fragment (16x32 f16): lane holds row l16; half selects K octets.
      const float* Xr = X + (size_t)ar * ldx + k0 + half * 8;
      const float4 x0 = *(const float4*)(Xr + 0);
      const float4 x1 = *(const float4*)(Xr + 4);
      const float4 x2 = *(const float4*)(Xr + 16);
      const float4 x3 = *(const float4*)(Xr + 20);
      v16h a;
      a[0] = (_Float16)x0.x; a[1] = (_Float16)x0.y; a[2] = (_Float16)x0.z; a[3] = (_Float16)x0.w;
      a[4] = (_Float16)x1.x; a[5] = (_Float16)x1.y; a[6] = (_Float16)x1.z; a[7] = (_Float16)x1.w;
      a[8] = (_Float16)x2.x; a[9] = (_Float16)x2.y; a[10] = (_Float16)x2.z; a[11] = (_Float16)x2.w;
      a[12] = (_Float16)x3.x; a[13] = (_Float16)x3.y; a[14] = (_Float16)x3.z; a[15] = (_Float16)x3.w;
      // B fragment (32x16 f16): lane holds column m; half selects K octets.
      const float* Wr = W + (size_t)mcl * ldw + k0 + half * 8;
      const float4 w0 = *(const float4*)(Wr + 0);
      const float4 w1 = *(const float4*)(Wr + 4);
      const float4 w2 = *(const float4*)(Wr + 16);
      const float4 w3 = *(const float4*)(Wr + 20);
      v16h b;
      b[0] = (_Float16)(w0.x * fmask); b[1] = (_Float16)(w0.y * fmask);
      b[2] = (_Float16)(w0.z * fmask); b[3] = (_Float16)(w0.w * fmask);
      b[4] = (_Float16)(w1.x * fmask); b[5] = (_Float16)(w1.y * fmask);
      b[6] = (_Float16)(w1.z * fmask); b[7] = (_Float16)(w1.w * fmask);
      b[8] = (_Float16)(w2.x * fmask); b[9] = (_Float16)(w2.y * fmask);
      b[10] = (_Float16)(w2.z * fmask); b[11] = (_Float16)(w2.w * fmask);
      b[12] = (_Float16)(w3.x * fmask); b[13] = (_Float16)(w3.y * fmask);
      b[14] = (_Float16)(w3.z * fmask); b[15] = (_Float16)(w3.w * fmask);
      c = __builtin_amdgcn_wmma_f32_16x16x32_f16(false, a, false, b,
                                                 (short)0, c, false, false);
    }
#pragma unroll
    for (int i = 0; i < 8; ++i) {
      int r = row0 + half * 8 + i;
      if (m < M) {
        float v0 = c[i];
        if (ACT == 1) v0 = fmaxf(v0, 0.f);
        else if (ACT == 2) v0 = fast_sigmoid(v0);
        Y[(size_t)r * ldy + m] = v0;
      }
    }
  }
}

// ---------------------------------------------------------------------------
// elementwise / graph kernels
// ---------------------------------------------------------------------------
__global__ void k_fill(float* p, float v, int n) {
  int i = blockIdx.x * blockDim.x + threadIdx.x;
  if (i < n) p[i] = v;
}
__global__ void k_add(float* y, const float* a, int n) {
  int i = blockIdx.x * blockDim.x + threadIdx.x;
  if (i < n) y[i] += a[i];
}
__global__ void k_scale_cols(float* y, const float* s, int n) {
  int i = blockIdx.x * blockDim.x + threadIdx.x;
  if (i < n) y[i] *= s[i % HDIM];
}
__global__ void k_fuse(const float* g, const float* a, const float* b, float* y, int n) {
  int i = blockIdx.x * blockDim.x + threadIdx.x;
  if (i < n) { float gg = g[i]; y[i] = gg * a[i] + (1.f - gg) * b[i]; }
}
// bn(x + pre_b) then act 0=none 1=relu 2=elu
template <int ACT>
__global__ void k_bn_act(const float* x, const float* pre_b, const float* g,
                         const float* b, const float* m, const float* v,
                         float* y, int total, int H) {
  int i = blockIdx.x * blockDim.x + threadIdx.x;
  if (i >= total) return;
  int f = i % H;
  float t = x[i];
  if (pre_b) t += pre_b[f];
  t = (t - m[f]) * rsqrtf(v[f] + 1e-5f) * g[f] + b[f];
  if (ACT == 1) t = fmaxf(t, 0.f);
  else if (ACT == 2) t = (t > 0.f) ? t : (__expf(t) - 1.f);
  y[i] = t;
}
__global__ void k_deg_edges(const int* dst, const float* ew, float* deg, int E) {
  int e = blockIdx.x * blockDim.x + threadIdx.x;
  if (e < E) atomAddF(&deg[dst[e]], ew[e]);
}
__global__ void k_dinv(float* deg, int n) {
  int i = blockIdx.x * blockDim.x + threadIdx.x;
  if (i < n) { float d = deg[i]; deg[i] = (d > 0.f) ? rsqrtf(fmaxf(d, 1e-12f)) : 0.f; }
}
// one wave32 per edge; lane covers 4 contiguous features (float4)
__global__ void k_gcn_scatter(const int* src, const int* dst, const float* ew,
                              const float* dinv, const float* xw, float* out, int E) {
  int e = blockIdx.x * (blockDim.x >> 5) + (threadIdx.x >> 5);
  if (e >= E) return;
  int lane = threadIdx.x & 31;
  int s = src[e], d = dst[e];
  float nrm = dinv[s] * ew[e] * dinv[d];
  const float* xr = xw + (size_t)s * HDIM + lane * 4;
  __builtin_prefetch(xr, 0, 0);
  float4 vx = *(const float4*)xr;
  float* o = out + (size_t)d * HDIM + lane * 4;
  atomAddF(o + 0, nrm * vx.x);
  atomAddF(o + 1, nrm * vx.y);
  atomAddF(o + 2, nrm * vx.z);
  atomAddF(o + 3, nrm * vx.w);
}
__global__ void k_gcn_self(const float* dinv, const float* xw, float* out, int n) {
  int node = blockIdx.x * (blockDim.x >> 5) + (threadIdx.x >> 5);
  if (node >= n) return;
  int lane = threadIdx.x & 31;
  float di = dinv[node];
  float nrm = di * 2.0f * di;
  const float4 vx = *(const float4*)(xw + (size_t)node * HDIM + lane * 4);
  float4* o = (float4*)(out + (size_t)node * HDIM + lane * 4);
  float4 t = *o;
  t.x += nrm * vx.x; t.y += nrm * vx.y; t.z += nrm * vx.z; t.w += nrm * vx.w;
  *o = t;
}
// per-edge gate MLP: few = ew * sigmoid(w2 . relu(ew*w1 + b1) + b2)
__global__ void k_edge_gate(const float* ew, const float* w1, const float* b1,
                            const float* w2, const float* b2, float* few, int E) {
  int e = blockIdx.x * blockDim.x + threadIdx.x;
  if (e >= E) return;
  float w = ew[e], acc = 0.f;
#pragma unroll
  for (int j = 0; j < 32; ++j) acc += w2[j] * fmaxf(w * w1[j] + b1[j], 0.f);
  few[e] = w * fast_sigmoid(acc + b2[0]);
}
__global__ void k_edot(const float* We, const float* attE, float* edot, int heads, int ch) {
  int h = threadIdx.x;
  if (h >= heads) return;
  float s = 0.f;
  for (int c = 0; c < ch; ++c) s += We[h * ch + c] * attE[h * ch + c];
  edot[h] = s;
}
__global__ void k_attn_nodes(const float* xw, const float* attS, const float* attD,
                             float* aS, float* aD, int n, int heads, int ch) {
  int t = blockIdx.x * blockDim.x + threadIdx.x;
  if (t >= n * heads) return;
  int node = t / heads, h = t % heads;
  const float* xr = xw + (size_t)node * HDIM + h * ch;
  float s = 0.f, d = 0.f;
  for (int c = 0; c < ch; ++c) { s += xr[c] * attS[h * ch + c]; d += xr[c] * attD[h * ch + c]; }
  aS[t] = s; aD[t] = d;
}
__global__ void k_cnt_sum(const int* dst, const float* few, float* cnt, float* sum, int E) {
  int e = blockIdx.x * blockDim.x + threadIdx.x;
  if (e >= E) return;
  int d = dst[e];
  atomAddF(&cnt[d], 1.f);
  atomAddF(&sum[d], few[e]);
}
__global__ void k_mean(const float* cnt, float* sum, int n) {
  int i = blockIdx.x * blockDim.x + threadIdx.x;
  if (i < n) sum[i] = sum[i] / fmaxf(cnt[i], 1.f);
}
__global__ void k_logit_max(const int* src, const int* dst, const float* few,
                            const float* meanea, const float* aS, const float* aD,
                            const float* edot, float* alpha, unsigned* amax,
                            int E, int n, int heads) {
  int idx = blockIdx.x * blockDim.x + threadIdx.x;
  if (idx >= E + n) return;
  int s, d; float ea;
  if (idx < E) { s = src[idx]; d = dst[idx]; ea = few[idx]; }
  else { s = d = idx - E; ea = meanea[s]; }
  for (int h = 0; h < heads; ++h) {
    float l = aS[s * heads + h] + aD[d * heads + h] + ea * edot[h];
    l = (l > 0.f) ? l : 0.2f * l;                 // leaky_relu(0.2)
    alpha[(size_t)idx * heads + h] = l;
    atomMaxU(&amax[d * heads + h], encf(l));
  }
}
__global__ void k_exp_sum(const int* src, const int* dst, const unsigned* amax,
                          float* alpha, float* asum, int E, int n, int heads) {
  int idx = blockIdx.x * blockDim.x + threadIdx.x;
  if (idx >= E + n) return;
  int d = (idx < E) ? dst[idx] : (idx - E);
  for (int h = 0; h < heads; ++h) {
    float al = __expf(alpha[(size_t)idx * heads + h] - decf(amax[d * heads + h]));
    alpha[(size_t)idx * heads + h] = al;
    atomAddF(&asum[d * heads + h], al);
  }
}
__global__ void k_attn_scatter(const int* src, const int* dst, const float* alpha,
                               const float* asum, const float* xw, float* out,
                               int E, int n, int heads, int ch) {
  int idx = blockIdx.x * (blockDim.x >> 5) + (threadIdx.x >> 5);
  if (idx >= E + n) return;
  int lane = threadIdx.x & 31;
  int s, d;
  if (idx < E) { s = src[idx]; d = dst[idx]; } else { s = d = idx - E; }
  int f0 = lane * 4;
  int h = f0 / ch;
  float w = alpha[(size_t)idx * heads + h] *
            __builtin_amdgcn_rcpf(asum[d * heads + h] + 1e-16f);
  const float4 vx = *(const float4*)(xw + (size_t)s * HDIM + f0);
  float* o = out + (size_t)d * HDIM + f0;
  atomAddF(o + 0, w * vx.x);
  atomAddF(o + 1, w * vx.y);
  atomAddF(o + 2, w * vx.z);
  atomAddF(o + 3, w * vx.w);
}

// ---------------------------------------------------------------------------
extern "C" void kernel_launch(void* const* d_in, const int* in_sizes, int n_in,
                              void* d_out, int out_size, void* d_ws, size_t ws_size,
                              hipStream_t stream) {
  (void)n_in; (void)out_size; (void)ws_size;
  const int E = in_sizes[2];
  const int N = in_sizes[0] / 64;
  const int NH = N * HDIM;
  const float* x   = (const float*)d_in[0];
  const int*   src = (const int*)d_in[1];
  const int*   dst = src + E;
  const float* ew  = (const float*)d_in[2];
#define F(k) ((const float*)d_in[(k)])
  // params['cls'] (final head, dout=2)
  const float *cls_b1=F(3), *cls_b2=F(4), *cls_b3=F(5);
  const float *cls_g=F(6), *cls_bb=F(7), *cls_m=F(8), *cls_v=F(9);
  const float *cls_w1=F(10), *cls_w2=F(11), *cls_w3=F(12);
  // params['gat']
  const float *gat_bg[3]={F(13),F(17),F(21)}, *gat_bbb[3]={F(14),F(18),F(22)};
  const float *gat_bm[3]={F(15),F(19),F(23)}, *gat_bv[3]={F(16),F(20),F(24)};
  const float *gc_b1=F(25), *gc_b2=F(26), *gc_b3=F(27);
  const float *gc_g=F(28), *gc_bb=F(29), *gc_m=F(30), *gc_v=F(31);
  const float *gc_w1=F(32), *gc_w2=F(33), *gc_w3=F(34);
  const float *cW[3]={F(35),F(41),F(47)}, *cWe[3]={F(36),F(42),F(48)};
  const float *cAd[3]={F(37),F(43),F(49)}, *cAe[3]={F(38),F(44),F(50)};
  const float *cAs[3]={F(39),F(45),F(51)}, *cBi[3]={F(40),F(46),F(52)};
  const float *ea_b1[3]={F(53),F(54),F(55)}, *ea_b2[3]={F(56),F(57),F(58)};
  const float *ea_w1[3]={F(59),F(60),F(61)}, *ea_w2[3]={F(62),F(63),F(64)};
  const float *gat_pb=F(65), *gat_pw=F(66), *symptom=F(67);
  const float *gate_b=F(68), *gate_w=F(69);
  // params['gcn']
  const float *n_bg[3]={F(70),F(74),F(78)}, *n_bb[3]={F(71),F(75),F(79)};
  const float *n_bm[3]={F(72),F(76),F(80)}, *n_bv[3]={F(73),F(77),F(81)};
  const float *nc_b1=F(82), *nc_b2=F(83), *nc_b3=F(84);
  const float *nc_g=F(85), *nc_bb=F(86), *nc_m=F(87), *nc_v=F(88);
  const float *nc_w1=F(89), *nc_w2=F(90), *nc_w3=F(91);
  const float *conv_b[3]={F(92),F(93),F(94)}, *conv_w[3]={F(95),F(96),F(97)};
  const float *mp_b1=F(98), *mp_b2=F(99), *mp_w1=F(100), *mp_w2=F(101);
  const float *gcn_pb=F(102), *gcn_pw=F(103);
  const float *res_b[2]={F(104),F(105)}, *res_w[2]={F(106),F(107)};
#undef F
  // workspace arena (256-B aligned carves keep float4 loads legal)
  char* wsp = (char*)d_ws; size_t off = 0;
  auto carve = [&](size_t bytes) -> void* {
    void* p = wsp + off; off += (bytes + 255) & ~(size_t)255; return p;
  };
  float* A    = (float*)carve((size_t)NH * 4);
  float* D    = (float*)carve((size_t)NH * 4);
  float* C    = (float*)carve((size_t)NH * 4);
  float* T    = (float*)carve((size_t)N * 32 * 4);
  float* gcnO = (float*)carve((size_t)NH * 4);
  float* gatO = (float*)carve((size_t)NH * 4);
  float* nb1  = (float*)carve((size_t)N * 4);
  float* nb2  = (float*)carve((size_t)N * 4);
  float* aS   = (float*)carve((size_t)N * 4 * 4);
  float* aD   = (float*)carve((size_t)N * 4 * 4);
  unsigned* amax = (unsigned*)carve((size_t)N * 4 * 4);
  float* asum = (float*)carve((size_t)N * 4 * 4);
  float* alpha= (float*)carve((size_t)(E + N) * 4 * 4);
  float* few  = (float*)carve((size_t)E * 4);
  float* edot = (float*)carve(64);

  const dim3 GB(256);                 // 8 waves: covers all 8 col-tiles of M=128
  const dim3 GG((N + 15) / 16);
  const dim3 EB(256);
#define GS(n) dim3(((n) + 255) / 256)
#define WGS(n) dim3(((n) + 7) / 8)    // wave-per-item kernels, 8 waves/block
  const int E2 = E + N;

  // ===================== GCN branch =====================
  k_gemm_wmma<0,0><<<GG, GB, 0, stream>>>(x, 64, gcn_pw, 64, gcn_pb, A, HDIM, N, HDIM, 64);
  for (int i = 0; i < 3; ++i) {
    k_gemm_wmma<0,0><<<GG, GB, 0, stream>>>(A, HDIM, conv_w[i], HDIM, nullptr, C, HDIM, N, HDIM, HDIM);
    k_fill<<<GS(N), EB, 0, stream>>>(nb1, 2.0f, N);                 // self-loop weight 2.0
    k_deg_edges<<<GS(E), EB, 0, stream>>>(dst, ew, nb1, E);
    k_dinv<<<GS(N), EB, 0, stream>>>(nb1, N);
    k_fill<<<GS(NH), EB, 0, stream>>>(D, 0.f, NH);
    k_gcn_scatter<<<WGS(E), EB, 0, stream>>>(src, dst, ew, nb1, C, D, E);
    k_gcn_self<<<WGS(N), EB, 0, stream>>>(nb1, C, D, N);
    k_bn_act<1><<<GS(NH), EB, 0, stream>>>(D, conv_b[i], n_bg[i], n_bb[i], n_bm[i], n_bv[i], D, NH, HDIM);
    if (i > 0)
      k_gemm_wmma<0,1><<<GG, GB, 0, stream>>>(A, HDIM, res_w[i-1], HDIM, res_b[i-1], D, HDIM, N, HDIM, HDIM);
    float* t = A; A = D; D = t;       // h <- hn (+res); old h becomes scratch
  }
  // pooled = lin(relu(lin(h))); h + pooled
  k_gemm_wmma<1,0><<<GG, GB, 0, stream>>>(A, HDIM, mp_w1, HDIM, mp_b1, C, 64, N, 64, HDIM);
  k_gemm_wmma<0,0><<<GG, GB, 0, stream>>>(C, 64, mp_w2, 64, mp_b2, D, HDIM, N, HDIM, 64);
  k_add<<<GS(NH), EB, 0, stream>>>(D, A, NH);
  // classifier_head(D) -> gcnO
  k_gemm_wmma<1,0><<<GG, GB, 0, stream>>>(D, HDIM, nc_w1, HDIM, nc_b1, C, 64, N, 64, HDIM);
  k_bn_act<0><<<GS(N*64), EB, 0, stream>>>(C, nullptr, nc_g, nc_bb, nc_m, nc_v, C, N*64, 64);
  k_gemm_wmma<1,0><<<GG, GB, 0, stream>>>(C, 64, nc_w2, 64, nc_b2, T, 32, N, 32, 64);
  k_gemm_wmma<0,0><<<GG, GB, 0, stream>>>(T, 32, nc_w3, 32, nc_b3, gcnO, HDIM, N, HDIM, 32);

  // ===================== GAT branch =====================
  k_gemm_wmma<0,0><<<GG, GB, 0, stream>>>(x, 64, gat_pw, 64, gat_pb, A, HDIM, N, HDIM, 64);
  for (int i = 0; i < 3; ++i) {
    const int heads = (i == 2) ? 1 : 4;
    const int ch = HDIM / heads;
    k_edge_gate<<<GS(E), EB, 0, stream>>>(ew, ea_w1[i], ea_b1[i], ea_w2[i], ea_b2[i], few, E);
    k_gemm_wmma<0,0><<<GG, GB, 0, stream>>>(A, HDIM, cW[i], HDIM, nullptr, C, HDIM, N, HDIM, HDIM);
    k_edot<<<1, heads, 0, stream>>>(cWe[i], cAe[i], edot, heads, ch);
    k_attn_nodes<<<GS(N*heads), EB, 0, stream>>>(C, cAs[i], cAd[i], aS, aD, N, heads, ch);
    k_fill<<<GS(N), EB, 0, stream>>>(nb1, 0.f, N);
    k_fill<<<GS(N), EB, 0, stream>>>(nb2, 0.f, N);
    k_cnt_sum<<<GS(E), EB, 0, stream>>>(dst, few, nb1, nb2, E);
    k_mean<<<GS(N), EB, 0, stream>>>(nb1, nb2, N);
    k_fill<<<GS(N*heads), EB, 0, stream>>>((float*)amax, 0.f, N*heads);
    k_logit_max<<<GS(E2), EB, 0, stream>>>(src, dst, few, nb2, aS, aD, edot, alpha, amax, E, N, heads);
    k_fill<<<GS(N*heads), EB, 0, stream>>>(asum, 0.f, N*heads);
    k_exp_sum<<<GS(E2), EB, 0, stream>>>(src, dst, amax, alpha, asum, E, N, heads);
    k_fill<<<GS(NH), EB, 0, stream>>>(D, 0.f, NH);
    k_attn_scatter<<<WGS(E2), EB, 0, stream>>>(src, dst, alpha, asum, C, D, E, N, heads, ch);
    k_bn_act<2><<<GS(NH), EB, 0, stream>>>(D, cBi[i], gat_bg[i], gat_bbb[i], gat_bm[i], gat_bv[i], A, NH, HDIM);
  }
  k_scale_cols<<<GS(NH), EB, 0, stream>>>(A, symptom, NH);
  // classifier_head(A) -> gatO
  k_gemm_wmma<1,0><<<GG, GB, 0, stream>>>(A, HDIM, gc_w1, HDIM, gc_b1, C, 64, N, 64, HDIM);
  k_bn_act<0><<<GS(N*64), EB, 0, stream>>>(C, nullptr, gc_g, gc_bb, gc_m, gc_v, C, N*64, 64);
  k_gemm_wmma<1,0><<<GG, GB, 0, stream>>>(C, 64, gc_w2, 64, gc_b2, T, 32, N, 32, 64);
  k_gemm_wmma<0,0><<<GG, GB, 0, stream>>>(T, 32, gc_w3, 32, gc_b3, gatO, HDIM, N, HDIM, 32);

  // ===================== fusion + final head =====================
  // gate = sigmoid([gcnO, gatO] @ gate_w^T + gate_b): two K=128 passes over K=256 weights
  k_gemm_wmma<0,0><<<GG, GB, 0, stream>>>(gcnO, HDIM, gate_w, 256, gate_b, C, HDIM, N, HDIM, HDIM);
  k_gemm_wmma<2,1><<<GG, GB, 0, stream>>>(gatO, HDIM, gate_w + HDIM, 256, nullptr, C, HDIM, N, HDIM, HDIM);
  k_fuse<<<GS(NH), EB, 0, stream>>>(C, gcnO, gatO, D, NH);
  k_gemm_wmma<1,0><<<GG, GB, 0, stream>>>(D, HDIM, cls_w1, HDIM, cls_b1, C, 64, N, 64, HDIM);
  k_bn_act<0><<<GS(N*64), EB, 0, stream>>>(C, nullptr, cls_g, cls_bb, cls_m, cls_v, C, N*64, 64);
  k_gemm_wmma<1,0><<<GG, GB, 0, stream>>>(C, 64, cls_w2, 64, cls_b2, T, 32, N, 32, 64);
  k_gemm_wmma<0,0><<<GG, GB, 0, stream>>>(T, 32, cls_w3, 32, cls_b3, (float*)d_out, 2, N, 2, 32);
#undef GS
#undef WGS
}
